// LoKr_55155970015578
// MI455X (gfx1250) — matrix operand order repaired and checked by reading, not compiled, not run on hardware
//
#include <hip/hip_runtime.h>
#include <hip/hip_bf16.h>

// LoKr: out_b = (1/16) * w1 @ X_b @ w2^T, X_b = reshape(x[b], 64x64)
// bf16x3 (hi/lo split) WMMA with f32 accumulation; memory-bound design.

typedef __attribute__((ext_vector_type(16))) __bf16 v16bf;
typedef __attribute__((ext_vector_type(8)))  __bf16 v8bf;
typedef __attribute__((ext_vector_type(8)))  float  v8f;

#define STR 72            // padded LDS row stride (elements) to spread banks
#define WAVES 4           // rows (waves) per block
#define TILEBF (64 * STR) // one 64x64 padded bf16 matrix in LDS

static __device__ inline v8f wmma_bf16(v16bf a, v16bf b, v8f c) {
    return __builtin_amdgcn_wmma_f32_16x16x32_bf16(false, a, false, b,
                                                   (short)0, c, false, false);
}

// 16 contiguous bf16 (32B, 16B-aligned) -> v16bf
static __device__ inline v16bf ld16(const __bf16* p) {
    v8bf a = *(const v8bf*)p;
    v8bf b = *(const v8bf*)(p + 8);
    return __builtin_shufflevector(a, b, 0,1,2,3,4,5,6,7,8,9,10,11,12,13,14,15);
}
// two 8-element chunks at p and p+16 (A-operand K-chunk pattern)
static __device__ inline v16bf ld8x2(const __bf16* p) {
    v8bf a = *(const v8bf*)p;
    v8bf b = *(const v8bf*)(p + 16);
    return __builtin_shufflevector(a, b, 0,1,2,3,4,5,6,7,8,9,10,11,12,13,14,15);
}

static __device__ inline void cvt16(const float* f, v16bf& h, v16bf& l) {
#pragma unroll
    for (int i = 0; i < 16; ++i) {
        __bf16 hb = (__bf16)f[i];
        h[i] = hb;
        l[i] = (__bf16)(f[i] - (float)hb);
    }
}

// ---------------- prep: ws = { w1_hi, w1_lo, w2_hi, w2_lo } --------------
// w1[l,m] = (1/16) * sum_r w1_a[r,l] * w1_b[m,r]   (scaling folded in)
// w2[k,n] =          sum_r w2_a[r,k] * w2_b[n,r]
__global__ void lokr_prep(const float* __restrict__ w1_a, const float* __restrict__ w1_b,
                          const float* __restrict__ w2_a, const float* __restrict__ w2_b,
                          __bf16* __restrict__ ws) {
    for (int i = threadIdx.x; i < 4096; i += 256) {
        int r = i >> 6, c = i & 63;
        float s1 = 0.f, s2 = 0.f;
#pragma unroll
        for (int j = 0; j < 16; ++j) {
            s1 += w1_a[j * 64 + r] * w1_b[c * 16 + j];
            s2 += w2_a[j * 64 + r] * w2_b[c * 16 + j];
        }
        s1 *= 0.0625f;
        __bf16 h1 = (__bf16)s1;
        ws[i]            = h1;
        ws[4096 + i]     = (__bf16)(s1 - (float)h1);
        __bf16 h2 = (__bf16)s2;
        ws[8192 + i]     = h2;
        ws[12288 + i]    = (__bf16)(s2 - (float)h2);
    }
}

// ---------------- main: one x-row per wave ----------------
__global__ void __launch_bounds__(WAVES * 32)
lokr_main(const float* __restrict__ x, const __bf16* __restrict__ ws,
          float* __restrict__ out) {
    extern __shared__ __align__(16) __bf16 smem[];
    __bf16* W1H = smem;                // (l,m) padded
    __bf16* W1L = W1H + TILEBF;
    __bf16* W2H = W1L + TILEBF;        // (k,n) padded
    __bf16* W2L = W2H + TILEBF;
    __bf16* YTB = W2L + TILEBF;        // per-wave YT hi/lo, (k,m) padded

    const int tid = threadIdx.x;
    // cooperative load of precomputed weights into padded LDS
    for (int i = tid; i < 4096; i += WAVES * 32) {
        int r = i >> 6, c = i & 63;
        W1H[r * STR + c] = ws[i];
        W1L[r * STR + c] = ws[4096 + i];
        W2H[r * STR + c] = ws[8192 + i];
        W2L[r * STR + c] = ws[12288 + i];
    }
    __syncthreads();

    const int wave = tid >> 5;
    const int lane = tid & 31;
    const int hl   = lane >> 4;   // half-wave select
    const int lm   = lane & 15;

    const int b = blockIdx.x * WAVES + wave;
    const float* xrow = x + (size_t)b * 4096;
    float*       orow = out + (size_t)b * 4096;
    __bf16* yth = YTB + wave * 2 * TILEBF;
    __bf16* ytl = yth + TILEBF;

    // ---- stage 1: YT = (X @ w2^T)^T, written to LDS as bf16 hi/lo ----
#pragma unroll
    for (int mt = 0; mt < 4; ++mt) {
        const int m = 16 * mt + lm;           // A row (same for both halves)
        v16bf ah[2], al[2];
#pragma unroll
        for (int nc = 0; nc < 2; ++nc) {      // K-window [32*nc, 32*nc+32)
            float f[16];
            const float* p = xrow + m * 64 + 32 * nc + hl * 8;
            *(float4*)&f[0]  = *(const float4*)(p);
            *(float4*)&f[4]  = *(const float4*)(p + 4);
            *(float4*)&f[8]  = *(const float4*)(p + 16);
            *(float4*)&f[12] = *(const float4*)(p + 20);
            cvt16(f, ah[nc], al[nc]);
        }
#pragma unroll
        for (int kt = 0; kt < 4; ++kt) {
            v8f acc = {};
            const int k = 16 * kt + lm;       // B column
#pragma unroll
            for (int nc = 0; nc < 2; ++nc) {
                const __bf16* ph = &W2H[k * STR + 32 * nc + hl * 8];
                const __bf16* pl = &W2L[k * STR + 32 * nc + hl * 8];
                v16bf bh = ld8x2(ph), bl = ld8x2(pl);
                acc = wmma_bf16(ah[nc], bh, acc);
                acc = wmma_bf16(ah[nc], bl, acc);
                acc = wmma_bf16(al[nc], bh, acc);
            }
            // split accumulator, store transposed: YT[k][m] (contiguous per lane)
            v8bf sh, sl;
#pragma unroll
            for (int v = 0; v < 8; ++v) {
                __bf16 hb = (__bf16)acc[v];
                sh[v] = hb;
                sl[v] = (__bf16)(acc[v] - (float)hb);
            }
            const int row = 16 * kt + lm, col = 16 * mt + hl * 8;
            *(v8bf*)&yth[row * STR + col] = sh;
            *(v8bf*)&ytl[row * STR + col] = sl;
        }
    }
    // YT is wave-private; same-wave LDS ops are in-order -> no barrier needed.

    // ---- stage 2: out_b = w1s @ Y ----
#pragma unroll
    for (int kt = 0; kt < 4; ++kt) {
        const int k = 16 * kt + lm;           // B column
        v16bf bh[2], bl[2];
#pragma unroll
        for (int mc = 0; mc < 2; ++mc) {
            bh[mc] = ld8x2(&yth[k * STR + 32 * mc + hl * 8]);
            bl[mc] = ld8x2(&ytl[k * STR + 32 * mc + hl * 8]);
        }
#pragma unroll
        for (int lt = 0; lt < 4; ++lt) {
            v8f acc = {};
            const int l = 16 * lt + lm;       // A row
#pragma unroll
            for (int mc = 0; mc < 2; ++mc) {
                const __bf16* ph = &W1H[l * STR + 32 * mc + hl * 8];
                const __bf16* pl = &W1L[l * STR + 32 * mc + hl * 8];
                v16bf av = ld8x2(ph), alo = ld8x2(pl);
                acc = wmma_bf16(av, bh[mc], acc);
                acc = wmma_bf16(av, bl[mc], acc);
                acc = wmma_bf16(alo, bh[mc], acc);
            }
#pragma unroll
            for (int v = 0; v < 8; ++v)
                orow[(16 * lt + hl * 8 + v) * 64 + 16 * kt + lm] = acc[v];
        }
    }
}

extern "C" void kernel_launch(void* const* d_in, const int* in_sizes, int n_in,
                              void* d_out, int out_size, void* d_ws, size_t ws_size,
                              hipStream_t stream) {
    (void)in_sizes; (void)n_in; (void)out_size; (void)ws_size;
    const float* x   = (const float*)d_in[0];
    const float* w1a = (const float*)d_in[1];
    const float* w1b = (const float*)d_in[2];
    const float* w2a = (const float*)d_in[3];
    const float* w2b = (const float*)d_in[4];
    __bf16* ws  = (__bf16*)d_ws;
    float*  out = (float*)d_out;

    hipLaunchKernelGGL(lokr_prep, dim3(1), dim3(256), 0, stream, w1a, w1b, w2a, w2b, ws);

    const size_t lds_bytes = (size_t)(4 + 2 * WAVES) * TILEBF * sizeof(__bf16); // 110,592 B
    hipLaunchKernelGGL(lokr_main, dim3(8192 / WAVES), dim3(WAVES * 32), lds_bytes, stream,
                       x, ws, out);
}
